// ContextualAggregationModule_71038759076481
// MI455X (gfx1250) — matrix operand order repaired and checked by reading, not compiled, not run on hardware
//
#include <hip/hip_runtime.h>
#include <math.h>
#include <stdint.h>

// Problem constants (fixed by the reference)
#define Bc   64
#define Nc   256
#define Dc   128
#define Hc   4
#define DKc  32
#define DFFc 2048
#define Lc   10
#define EPSc 1e-5f

typedef __attribute__((ext_vector_type(16))) _Float16 v16h;
typedef __attribute__((ext_vector_type(8)))  _Float16 v8h;
typedef __attribute__((ext_vector_type(4)))  _Float16 v4h;
typedef __attribute__((ext_vector_type(8)))  float    v8f;

union Frag16 { v16h v; v8h p[2]; _Float16 h[16]; };

#define WMMA_F16(a, b, c) __builtin_amdgcn_wmma_f32_16x16x32_f16( \
    false, (a).v, false, (b).v, (short)0, (c), false, false)

// A fragment (16x32, row-major f16, lda = K): per lane (half=lane>>4, m=lane&15)
// one contiguous 16-element run at K = k0+half*16.
__device__ __forceinline__ void load_a16(Frag16& f, const _Float16* __restrict__ row,
                                         int k0, int half) {
  f.p[0] = *reinterpret_cast<const v8h*>(row + k0 + half * 16);
  f.p[1] = *reinterpret_cast<const v8h*>(row + k0 + half * 16 + 8);
}

// B fragment for A @ W^T (W row-major N x K): one contiguous 16-half run at
// K = k0 + half*16 for column n = lane&15.
__device__ __forceinline__ void load_b16(Frag16& f, const _Float16* __restrict__ row,
                                         int k0, int half) {
  f.p[0] = *reinterpret_cast<const v8h*>(row + k0 + half * 16);
  f.p[1] = *reinterpret_cast<const v8h*>(row + k0 + half * 16 + 8);
}

// LDS byte address of a __shared__ object = low 32 bits of its flat address
// (AS3 -> flat cast keeps the LDS offset in addr[31:0]).
__device__ __forceinline__ uint32_t lds_addr32(const void* p) {
  return (uint32_t)(uintptr_t)p;
}

// ---- Generic f16 NT GEMM, 32x32 per wave: C = act(A[MxK] @ W[NxK]^T + bias) --
// The 8 waves of a block share one 32-row W slab; it is staged into LDS via
// GLOBAL_LOAD_ASYNC_TO_LDS_B64 (ASYNCcnt) each K-step, then read as ds_load_b128.
// MODE: 0 = f32 row-major, 1 = f16 row-major, 2 = f16 (B,H,N,DK) qkv-permute,
//       3 = f16 (B,H,DK,N) transposed-V store.
template <int RELU, int MODE>
__global__ void wmma_gemm16(const _Float16* __restrict__ A, const _Float16* __restrict__ W,
                            const float* __restrict__ bias, void* __restrict__ Cv,
                            int M, int N, int K) {
  __shared__ _Float16 ldsW[32 * 32];           // 32 N-rows x 32 K-halves = 2 KB
  int lane = threadIdx.x & 31;
  int wave = threadIdx.x >> 5;
  int m0 = blockIdx.x * 256 + wave * 32;
  int n0 = blockIdx.y * 32;
  if (m0 >= M) return;                         // grid divides exactly; never taken
  int half = lane >> 4;
  int ln   = lane & 15;
  const _Float16* A0 = A + (size_t)(m0 + ln) * K;
  const _Float16* A1 = A + (size_t)(m0 + 16 + ln) * K;
  // cooperative async-copy assignment: 256 threads x 8 bytes = 2 KB slab
  int crow = threadIdx.x >> 3;                 // 0..31  (N row within slab)
  int cseg = threadIdx.x & 7;                  // 0..7   (4-half segment)
  const _Float16* wsrc_base = W + (size_t)(n0 + crow) * K + cseg * 4;
  uint32_t wdst = lds_addr32(&ldsW[crow * 32 + cseg * 4]);
  v8f acc[2][2] = {};
  for (int k0 = 0; k0 < K; k0 += 32) {
    __syncthreads();                           // LDS WAR vs previous iteration
    asm volatile("global_load_async_to_lds_b64 %0, %1, off"
                 :: "v"(wdst), "v"(wsrc_base + k0) : "memory");
    if (k0 + 32 < K)                           // global_prefetch_b8 of next A slab
      __builtin_prefetch(A0 + k0 + 32, 0, 1);
    asm volatile("s_wait_asynccnt 0x0" ::: "memory");
    __syncthreads();                           // slab visible to all 8 waves
    Frag16 a0, a1, b0, b1;
    load_a16(a0, A0, k0, half);
    load_a16(a1, A1, k0, half);
    b0.p[0] = *reinterpret_cast<const v8h*>(&ldsW[ln * 32 + half * 16]);
    b0.p[1] = *reinterpret_cast<const v8h*>(&ldsW[ln * 32 + half * 16 + 8]);
    b1.p[0] = *reinterpret_cast<const v8h*>(&ldsW[(16 + ln) * 32 + half * 16]);
    b1.p[1] = *reinterpret_cast<const v8h*>(&ldsW[(16 + ln) * 32 + half * 16 + 8]);
    acc[0][0] = WMMA_F16(a0, b0, acc[0][0]);
    acc[0][1] = WMMA_F16(a0, b1, acc[0][1]);
    acc[1][0] = WMMA_F16(a1, b0, acc[1][0]);
    acc[1][1] = WMMA_F16(a1, b1, acc[1][1]);
  }
#pragma unroll
  for (int tm = 0; tm < 2; ++tm) {
#pragma unroll
    for (int tn = 0; tn < 2; ++tn) {
      int col = n0 + tn * 16 + ln;
      float bv = bias[col];
#pragma unroll
      for (int r = 0; r < 8; ++r) {
        int row = m0 + tm * 16 + half * 8 + r;
        float v = acc[tm][tn][r] + bv;
        if (RELU) v = fmaxf(v, 0.0f);
        if (MODE == 0) {
          ((float*)Cv)[(size_t)row * N + col] = v;
        } else if (MODE == 1) {
          ((_Float16*)Cv)[(size_t)row * N + col] = (_Float16)v;
        } else {
          int b_ = row >> 8, i_ = row & 255;     // seq = 256
          int h_ = col >> 5, d_ = col & 31;      // DK  = 32
          if (MODE == 2)
            ((_Float16*)Cv)[((size_t)(b_ * Hc + h_) * Nc + i_) * DKc + d_] = (_Float16)v;
          else  // MODE == 3: V^T layout (B,H,DK,N)
            ((_Float16*)Cv)[((size_t)(b_ * Hc + h_) * DKc + d_) * Nc + i_] = (_Float16)v;
        }
      }
    }
  }
}

// ---- Batched scores: att[bh] = Q[bh] @ K[bh]^T / sqrt(DK) + knn_bias[b] ------
__global__ void wmma_scores16(const _Float16* __restrict__ Q, const _Float16* __restrict__ Kt,
                              const float* __restrict__ biasW, float* __restrict__ att) {
  int lane = threadIdx.x & 31;
  int wave = threadIdx.x >> 5;
  int bh = blockIdx.z;
  int b  = bh >> 2;
  const _Float16* Ab = Q  + (size_t)bh * Nc * DKc;
  const _Float16* Bb = Kt + (size_t)bh * Nc * DKc;
  const float* wb = biasW + (size_t)b * Nc * Nc;
  float* Cb = att + (size_t)bh * Nc * Nc;
  int m0 = wave * 32;
  int n0 = blockIdx.y * 32;
  int half = lane >> 4, ln = lane & 15;
  const _Float16* A0 = Ab + (size_t)(m0 + ln) * DKc;
  const _Float16* A1 = Ab + (size_t)(m0 + 16 + ln) * DKc;
  const _Float16* B0 = Bb + (size_t)(n0 + ln) * DKc;
  const _Float16* B1 = Bb + (size_t)(n0 + 16 + ln) * DKc;
  Frag16 a0, a1, b0, b1;
  load_a16(a0, A0, 0, half);
  load_a16(a1, A1, 0, half);
  load_b16(b0, B0, 0, half);
  load_b16(b1, B1, 0, half);
  v8f z = {};
  v8f acc[2][2];
  acc[0][0] = WMMA_F16(a0, b0, z);
  acc[0][1] = WMMA_F16(a0, b1, z);
  acc[1][0] = WMMA_F16(a1, b0, z);
  acc[1][1] = WMMA_F16(a1, b1, z);
  const float rs = 0.17677669529663687f;     // 1/sqrt(32)
#pragma unroll
  for (int tm = 0; tm < 2; ++tm)
#pragma unroll
    for (int tn = 0; tn < 2; ++tn)
#pragma unroll
      for (int r = 0; r < 8; ++r) {
        int row = m0 + tm * 16 + half * 8 + r;
        int col = n0 + tn * 16 + ln;
        Cb[(size_t)row * Nc + col] = acc[tm][tn][r] * rs + wb[(size_t)row * Nc + col];
      }
}

// ---- Batched att @ V (V pre-transposed f16), un-permute heads to (B*N, D) ----
__global__ void wmma_av16(const _Float16* __restrict__ atth, const _Float16* __restrict__ Vt,
                          _Float16* __restrict__ proj) {
  int lane = threadIdx.x & 31;
  int wave = threadIdx.x >> 5;
  int bh = blockIdx.z;
  int b  = bh >> 2, hh = bh & 3;
  const _Float16* Ab = atth + (size_t)bh * Nc * Nc;   // 256x256 row-major
  const _Float16* Bt = Vt   + (size_t)bh * Nc * DKc;  // DK x 256 row-major
  int m0 = wave * 32;
  int half = lane >> 4, ln = lane & 15;
  const _Float16* A0 = Ab + (size_t)(m0 + ln) * Nc;
  const _Float16* A1 = Ab + (size_t)(m0 + 16 + ln) * Nc;
  const _Float16* B0 = Bt + (size_t)ln * Nc;          // d = ln
  const _Float16* B1 = Bt + (size_t)(16 + ln) * Nc;   // d = 16+ln
  v8f acc[2][2] = {};
  for (int k0 = 0; k0 < Nc; k0 += 32) {
    if (k0 + 32 < Nc) {
      __builtin_prefetch(A0 + k0 + 32, 0, 1);
      __builtin_prefetch(B0 + k0 + 32, 0, 1);
    }
    Frag16 a0, a1, b0, b1;
    load_a16(a0, A0, k0, half);
    load_a16(a1, A1, k0, half);
    load_b16(b0, B0, k0, half);
    load_b16(b1, B1, k0, half);
    acc[0][0] = WMMA_F16(a0, b0, acc[0][0]);
    acc[0][1] = WMMA_F16(a0, b1, acc[0][1]);
    acc[1][0] = WMMA_F16(a1, b0, acc[1][0]);
    acc[1][1] = WMMA_F16(a1, b1, acc[1][1]);
  }
#pragma unroll
  for (int tm = 0; tm < 2; ++tm)
#pragma unroll
    for (int tn = 0; tn < 2; ++tn)
#pragma unroll
      for (int r = 0; r < 8; ++r) {
        int row = m0 + tm * 16 + half * 8 + r;
        int col = tn * 16 + ln;
        proj[((size_t)(b * Nc + row)) * Dc + hh * DKc + col] = (_Float16)acc[tm][tn][r];
      }
}

// ---- Softmax over rows of 256 (one wave per row), writes f16 -----------------
__global__ void softmax16(const float* __restrict__ att, _Float16* __restrict__ atth,
                          int rows) {
  int row = blockIdx.x * 8 + (threadIdx.x >> 5);
  if (row >= rows) return;
  int lane = threadIdx.x & 31;
  const float* p = att + (size_t)row * Nc;
  _Float16* q = atth + (size_t)row * Nc;
  float vals[8];
  float m = -INFINITY;
#pragma unroll
  for (int j = 0; j < 8; ++j) { vals[j] = p[lane + 32 * j]; m = fmaxf(m, vals[j]); }
#pragma unroll
  for (int off = 16; off > 0; off >>= 1) m = fmaxf(m, __shfl_xor(m, off, 32));
  float s = 0.0f;
#pragma unroll
  for (int j = 0; j < 8; ++j) { vals[j] = __expf(vals[j] - m); s += vals[j]; }
#pragma unroll
  for (int off = 16; off > 0; off >>= 1) s += __shfl_xor(s, off, 32);
  float inv = 1.0f / s;
#pragma unroll
  for (int j = 0; j < 8; ++j) q[lane + 32 * j] = (_Float16)(vals[j] * inv);
}

// ---- out = LayerNorm(x + res) * g + b ; optional f16 mirror for next GEMM ----
template <int W16>
__global__ void add_layernorm(const float* __restrict__ x, const float* __restrict__ r,
                              const float* __restrict__ g, const float* __restrict__ bb,
                              float* __restrict__ out, _Float16* __restrict__ out16,
                              int rows) {
  int row = blockIdx.x * 8 + (threadIdx.x >> 5);
  if (row >= rows) return;
  int lane = threadIdx.x & 31;
  const float* px = x + (size_t)row * Dc;
  const float* pr = r + (size_t)row * Dc;
  float v[4];
  float s = 0.0f;
#pragma unroll
  for (int j = 0; j < 4; ++j) { v[j] = px[lane + 32 * j] + pr[lane + 32 * j]; s += v[j]; }
#pragma unroll
  for (int off = 16; off > 0; off >>= 1) s += __shfl_xor(s, off, 32);
  float mu = s * (1.0f / Dc);
  float q = 0.0f;
#pragma unroll
  for (int j = 0; j < 4; ++j) { float d = v[j] - mu; q += d * d; }
#pragma unroll
  for (int off = 16; off > 0; off >>= 1) q += __shfl_xor(q, off, 32);
  float rstd = rsqrtf(q * (1.0f / Dc) + EPSc);
#pragma unroll
  for (int j = 0; j < 4; ++j) {
    int c = lane + 32 * j;
    float val = g[c] * (v[j] - mu) * rstd + bb[c];
    out[(size_t)row * Dc + c] = val;
    if (W16) out16[(size_t)row * Dc + c] = (_Float16)val;
  }
}

// ---- fp32 -> f16 bulk convert (weights / input features) ---------------------
__global__ void cvt16(const float* __restrict__ in, _Float16* __restrict__ out, int n) {
  int i = (blockIdx.x * blockDim.x + threadIdx.x) * 4;
  if (i >= n) return;
  float4 v = *reinterpret_cast<const float4*>(in + i);
  v4h h;
  h.x = (_Float16)v.x; h.y = (_Float16)v.y; h.z = (_Float16)v.z; h.w = (_Float16)v.w;
  *reinterpret_cast<v4h*>(out + i) = h;
}

// ---- Spatial proximity: dist[b,i,j] = ||src[b,j]-tar[b,i]||, inv = 1/(d+.01) -
__global__ void dist_inv_kernel(const float* __restrict__ src, const float* __restrict__ tar,
                                float* __restrict__ dist, float* __restrict__ inv) {
  int bi = blockIdx.x;               // b*256 + i
  int b  = bi >> 8;
  int j  = threadIdx.x;
  const float* t = tar + (size_t)bi * 3;
  const float* s = src + ((size_t)b * Nc + j) * 3;
  float dx = s[0] - t[0], dy = s[1] - t[1], dz = s[2] - t[2];
  float d = sqrtf(dx * dx + dy * dy + dz * dz);
  dist[(size_t)bi * Nc + j] = d;
  inv[(size_t)bi * Nc + j]  = 1.0f / (d + 0.01f);
}

// ---- colsum[b,j] = sum_i inv[b,i,j] (normalize over axis=2 as in torch) ------
__global__ void colsum_kernel(const float* __restrict__ inv, float* __restrict__ cs) {
  int idx = blockIdx.x * blockDim.x + threadIdx.x;   // b*256 + j
  int b = idx >> 8, j = idx & 255;
  float s = 0.0f;
  for (int i = 0; i < Nc; ++i) s += inv[((size_t)b * Nc + i) * Nc + j];
  cs[idx] = s;
}

// ---- rank-based top-L selection + bias build (LDS row, index tie-break) ------
__global__ void knn_bias_kernel(const float* __restrict__ dist, const float* __restrict__ inv,
                                const float* __restrict__ cs, float* __restrict__ biasW) {
  __shared__ float rowd[Nc];
  int bi = blockIdx.x;
  int j  = threadIdx.x;
  int b  = bi >> 8;
  float d = dist[(size_t)bi * Nc + j];
  rowd[j] = d;
  __syncthreads();
  int rank = 0;
  for (int t = 0; t < Nc; ++t) {
    float o = rowd[t];
    rank += (o < d) || (o == d && t < j);
  }
  float w = inv[(size_t)bi * Nc + j] / cs[b * Nc + j];
  biasW[(size_t)bi * Nc + j] = (rank < Lc) ? w : -INFINITY;
}

// ---- offsets: imm = centers + relu_hidden(f16) @ off_W2^T + off_b2 (Nout=3) --
__global__ void off_proj_kernel(const _Float16* __restrict__ hdn, const float* __restrict__ W2,
                                const float* __restrict__ b2, const float* __restrict__ centers,
                                float* __restrict__ imm) {
  int idx = blockIdx.x * blockDim.x + threadIdx.x;   // m*3 + c
  if (idx >= Bc * Nc * 3) return;
  int m = idx / 3, c = idx % 3;
  const _Float16* hp = hdn + (size_t)m * Dc;
  const float*    wp = W2  + (size_t)c * Dc;
  float s = 0.0f;
  for (int k = 0; k < Dc; ++k) s += (float)hp[k] * wp[k];
  imm[idx] = centers[idx] + s + b2[c];
}

// -----------------------------------------------------------------------------
extern "C" void kernel_launch(void* const* d_in, const int* in_sizes, int n_in,
                              void* d_out, int out_size, void* d_ws, size_t ws_size,
                              hipStream_t stream) {
  const float* centers    = (const float*)d_in[0];
  const float* features   = (const float*)d_in[1];
  const float* o_centers  = (const float*)d_in[2];
  const float* o_features = (const float*)d_in[3];
  const float* sa_Wq = (const float*)d_in[4];  const float* sa_bq = (const float*)d_in[5];
  const float* sa_Wk = (const float*)d_in[6];  const float* sa_bk = (const float*)d_in[7];
  const float* sa_Wv = (const float*)d_in[8];  const float* sa_bv = (const float*)d_in[9];
  const float* sa_Wo = (const float*)d_in[10]; const float* sa_bo = (const float*)d_in[11];
  const float* sa_g  = (const float*)d_in[12]; const float* sa_b  = (const float*)d_in[13];
  const float* ca_Wq = (const float*)d_in[14]; const float* ca_bq = (const float*)d_in[15];
  const float* ca_Wk = (const float*)d_in[16]; const float* ca_bk = (const float*)d_in[17];
  const float* ca_Wv = (const float*)d_in[18]; const float* ca_bv = (const float*)d_in[19];
  const float* ca_Wo = (const float*)d_in[20]; const float* ca_bo = (const float*)d_in[21];
  const float* ca_g  = (const float*)d_in[22]; const float* ca_b  = (const float*)d_in[23];
  const float* off_W1 = (const float*)d_in[24]; const float* off_b1 = (const float*)d_in[25];
  const float* off_W2 = (const float*)d_in[26]; const float* off_b2 = (const float*)d_in[27];
  const float* ff_W1 = (const float*)d_in[28]; const float* ff_b1 = (const float*)d_in[29];
  const float* ff_W2 = (const float*)d_in[30]; const float* ff_b2 = (const float*)d_in[31];
  const float* ff_g  = (const float*)d_in[32]; const float* ff_b  = (const float*)d_in[33];
  (void)in_sizes; (void)n_in; (void)out_size; (void)ws_size;

  char* base = (char*)d_ws;
  size_t off = 0;
  auto allocf = [&](size_t n) {
    float* p = (float*)(base + off); off = (off + n * 4 + 15) & ~(size_t)15; return p;
  };
  auto alloch = [&](size_t n) {
    _Float16* p = (_Float16*)(base + off); off = (off + n * 2 + 15) & ~(size_t)15; return p;
  };
  const size_t BN  = (size_t)Bc * Nc;          // 16384
  const size_t BND = BN * Dc;                  // 2.1M
  const size_t BNN = BN * Nc;                  // 4.19M
  // f32 scratch
  float* dist   = allocf(BNN);
  float* inv    = allocf(BNN);
  float* cs     = allocf(BN);
  float* biasW  = allocf(BNN);
  float* att    = allocf((size_t)Hc * BNN);    // 16.8M
  float* obuf   = allocf(BND);
  float* feats1 = allocf(BND);
  float* feats2 = allocf(BND);
  // f16 scratch
  _Float16* feat16  = alloch(BND);
  _Float16* ofeat16 = alloch(BND);
  _Float16* q16     = alloch(BND);
  _Float16* k16     = alloch(BND);
  _Float16* vt16    = alloch(BND);
  _Float16* atth    = alloch((size_t)Hc * BNN);
  _Float16* proj16  = alloch(BND);
  _Float16* f1_16   = alloch(BND);
  _Float16* f2_16   = alloch(BND);
  _Float16* hdn16   = alloch(BND);
  _Float16* ffh16   = alloch(BN * DFFc);       // 33.6M halves
  _Float16* sa_Wq16 = alloch(Dc * Dc); _Float16* sa_Wk16 = alloch(Dc * Dc);
  _Float16* sa_Wv16 = alloch(Dc * Dc); _Float16* sa_Wo16 = alloch(Dc * Dc);
  _Float16* ca_Wq16 = alloch(Dc * Dc); _Float16* ca_Wk16 = alloch(Dc * Dc);
  _Float16* ca_Wv16 = alloch(Dc * Dc); _Float16* ca_Wo16 = alloch(Dc * Dc);
  _Float16* offW116 = alloch(Dc * Dc);
  _Float16* ffW116  = alloch((size_t)DFFc * Dc);
  _Float16* ffW216  = alloch((size_t)Dc * DFFc);

  const int M = (int)BN;
  dim3 blk(256);
  auto cvt = [&](const float* in, _Float16* out, int n) {
    cvt16<<<(n / 4 + 255) / 256, blk, 0, stream>>>(in, out, n);
  };

  // ---- one-time f16 conversions ----
  cvt(features,   feat16,  (int)BND);
  cvt(o_features, ofeat16, (int)BND);
  cvt(sa_Wq, sa_Wq16, Dc * Dc); cvt(sa_Wk, sa_Wk16, Dc * Dc);
  cvt(sa_Wv, sa_Wv16, Dc * Dc); cvt(sa_Wo, sa_Wo16, Dc * Dc);
  cvt(ca_Wq, ca_Wq16, Dc * Dc); cvt(ca_Wk, ca_Wk16, Dc * Dc);
  cvt(ca_Wv, ca_Wv16, Dc * Dc); cvt(ca_Wo, ca_Wo16, Dc * Dc);
  cvt(off_W1, offW116, Dc * Dc);
  cvt(ff_W1, ffW116, DFFc * Dc);
  cvt(ff_W2, ffW216, Dc * DFFc);

  dim3 gProj(M / 256, Dc / 32);        // (64, 4)
  dim3 gFF1(M / 256, DFFc / 32);       // (64, 64)
  dim3 gScore(1, Nc / 32, Bc * Hc);    // (1, 8, 256)
  dim3 gAV(1, 1, Bc * Hc);

  // ---- self-attention proximity bias ----
  dist_inv_kernel<<<M, blk, 0, stream>>>(centers, centers, dist, inv);
  colsum_kernel<<<M / 256, blk, 0, stream>>>(inv, cs);
  knn_bias_kernel<<<M, blk, 0, stream>>>(dist, inv, cs, biasW);

  // ---- self-attention ----
  wmma_gemm16<0, 2><<<gProj, blk, 0, stream>>>(feat16, sa_Wq16, sa_bq, q16, M, Dc, Dc);
  wmma_gemm16<0, 2><<<gProj, blk, 0, stream>>>(feat16, sa_Wk16, sa_bk, k16, M, Dc, Dc);
  wmma_gemm16<0, 3><<<gProj, blk, 0, stream>>>(feat16, sa_Wv16, sa_bv, vt16, M, Dc, Dc);
  wmma_scores16<<<gScore, blk, 0, stream>>>(q16, k16, biasW, att);
  softmax16<<<(Bc * Hc * Nc) / 8, blk, 0, stream>>>(att, atth, Bc * Hc * Nc);
  wmma_av16<<<gAV, blk, 0, stream>>>(atth, vt16, proj16);
  wmma_gemm16<0, 0><<<gProj, blk, 0, stream>>>(proj16, sa_Wo16, sa_bo, obuf, M, Dc, Dc);
  add_layernorm<1><<<M / 8, blk, 0, stream>>>(features, obuf, sa_g, sa_b, feats1, f1_16, M);

  // ---- offset FFN -> imm (first output) ----
  wmma_gemm16<1, 1><<<gProj, blk, 0, stream>>>(f1_16, offW116, off_b1, hdn16, M, Dc, Dc);
  float* imm = (float*)d_out;
  off_proj_kernel<<<(M * 3 + 255) / 256, blk, 0, stream>>>(hdn16, off_W2, off_b2, centers, imm);

  // ---- cross-attention proximity bias ----
  dist_inv_kernel<<<M, blk, 0, stream>>>(imm, o_centers, dist, inv);
  colsum_kernel<<<M / 256, blk, 0, stream>>>(inv, cs);
  knn_bias_kernel<<<M, blk, 0, stream>>>(dist, inv, cs, biasW);

  // ---- cross-attention ----
  wmma_gemm16<0, 2><<<gProj, blk, 0, stream>>>(f1_16, ca_Wq16, ca_bq, q16, M, Dc, Dc);
  wmma_gemm16<0, 2><<<gProj, blk, 0, stream>>>(ofeat16, ca_Wk16, ca_bk, k16, M, Dc, Dc);
  wmma_gemm16<0, 3><<<gProj, blk, 0, stream>>>(ofeat16, ca_Wv16, ca_bv, vt16, M, Dc, Dc);
  wmma_scores16<<<gScore, blk, 0, stream>>>(q16, k16, biasW, att);
  softmax16<<<(Bc * Hc * Nc) / 8, blk, 0, stream>>>(att, atth, Bc * Hc * Nc);
  wmma_av16<<<gAV, blk, 0, stream>>>(atth, vt16, proj16);
  wmma_gemm16<0, 0><<<gProj, blk, 0, stream>>>(proj16, ca_Wo16, ca_bo, obuf, M, Dc, Dc);
  add_layernorm<1><<<M / 8, blk, 0, stream>>>(feats1, obuf, ca_g, ca_b, feats2, f2_16, M);

  // ---- main FFN -> feats (second output) ----
  wmma_gemm16<1, 1><<<gFF1, blk, 0, stream>>>(f2_16, ffW116, ff_b1, ffh16, M, DFFc, Dc);
  wmma_gemm16<0, 0><<<gProj, blk, 0, stream>>>(ffh16, ffW216, ff_b2, obuf, M, Dc, DFFc);
  float* out_feats = (float*)d_out + BN * 3;
  add_layernorm<0><<<M / 8, blk, 0, stream>>>(feats2, obuf, ff_g, ff_b, out_feats, nullptr, M);
}
